// LunarCausalAttention_25391846654053
// MI455X (gfx1250) — matrix operand order repaired and checked by reading, not compiled, not run on hardware
//
#include <hip/hip_runtime.h>

// ---------------------------------------------------------------------------
// LunarCausalAttention for MI455X (gfx1250, wave32, WMMA).
//
// Pipeline (all on `stream`):
//   1. f32->bf16 casts of query + 4 weight matrices  (everything L2-resident)
//   2. pq = (pquery @ Wpq^T + bpq) * scale           (tiny, f32 VALU)
//   3. kp/q/kv = (X @ W^T + b) * s   via v_wmma_f32_16x16x32_bf16
//   4. pattn = kp @ pq^T per head, softplus via log2(1+2^x)
//   5. fused chunked causal linear attention (T=32): WMMA intra-chunk,
//      f32 LDS state for the prefix sums, softmax over 32
//   6. out = attn @ Wo^T + bo  (WMMA, f32 output)
//
// Workspace requirement: ~55 MB.
// ---------------------------------------------------------------------------

typedef __bf16 bf16_t;
typedef __bf16 v16bf __attribute__((ext_vector_type(16)));
typedef __bf16 v8bf  __attribute__((ext_vector_type(8)));
typedef float v8f    __attribute__((ext_vector_type(8)));

#define EMB   1024
#define TGT   2048
#define BSZ   2
#define NTOK  (TGT*BSZ)   // 4096
#define NHEAD 16
#define HD    64          // head dim
#define NBH   (BSZ*NHEAD) // 32
#define PLEN  32
#define SCALING 0.125f    // 64^-0.5

// ---------------- WMMA fragment helpers (layouts per CDNA5 ISA 7.12.2) -----

// A-matrix 16x32 bf16, row-major source with row stride `ld` (elements).
// lane(0..15): K 0..7 & 16..23 of row M=lane ; lane(16..31): K 8..15 & 24..31.
__device__ __forceinline__ v16bf load_a_frag(const bf16_t* src, int ld) {
  int lane = threadIdx.x & 31;
  int m = lane & 15, half = lane >> 4;
  const bf16_t* p = src + (size_t)m * ld + half * 8;
  union { v16bf v; v8bf h[2]; } u;
  u.h[0] = *(const v8bf*)(p);
  u.h[1] = *(const v8bf*)(p + 16);
  return u.v;
}

// B-matrix 32x16 bf16 where logical B = X^T for row-major X (row stride ld):
// B column n = X row n. lane(0..15): K 0..7 & 16..23 ; lane(16..31): K 8..15 & 24..31.
__device__ __forceinline__ v16bf load_bT_frag(const bf16_t* x, int ld) {
  int lane = threadIdx.x & 31;
  int n = lane & 15, half = lane >> 4;
  const bf16_t* p = x + (size_t)n * ld + half * 8;
  union { v16bf v; v8bf h[2]; } u;
  u.h[0] = *(const v8bf*)(p);
  u.h[1] = *(const v8bf*)(p + 16);
  return u.v;
}

__device__ __forceinline__ v8f wmma_bf16(v16bf a, v16bf b, v8f c) {
  return __builtin_amdgcn_wmma_f32_16x16x32_bf16(
      /*neg_a=*/false, a, /*neg_b=*/false, b,
      /*c_mod=*/(short)0, c, /*reuse_a=*/false, /*reuse_b=*/false);
}

// ---------------- tiny utility kernels -------------------------------------

__global__ void cvt_f32_bf16(const float* __restrict__ s, bf16_t* __restrict__ d, int n) {
  int i = blockIdx.x * blockDim.x + threadIdx.x;
  if (i < n) d[i] = (bf16_t)s[i];
}

// pq = (pquery @ Wpq^T + bpq) * SCALING  -> bf16 (64 x 1024), f32 math.
__global__ void pq_proj(const float* __restrict__ x, const float* __restrict__ W,
                        const float* __restrict__ b, bf16_t* __restrict__ out) {
  int col = blockIdx.x * blockDim.x + threadIdx.x;  // 0..1023
  int row = blockIdx.y;                             // 0..63
  const float* xr = x + (size_t)row * EMB;
  const float* wr = W + (size_t)col * EMB;
  float acc = 0.f;
  for (int k = 0; k < EMB; ++k) acc = fmaf(xr[k], wr[k], acc);
  out[(size_t)row * EMB + col] = (bf16_t)((acc + b[col]) * SCALING);
}

// ---------------- main token GEMM: Y = (X @ W^T + b) * scale ---------------
// X (M x 1024) bf16 row-major, W (1024 x 1024) bf16 row-major (torch Linear
// layout => B-fragments load W rows directly, no transpose needed).
// Block = 256 thr (8 waves), tile 128(M) x 64(N); wave does 32x32 (2x2 tiles).

template <bool F32OUT>
__global__ void __launch_bounds__(256)
gemm_xwT(const bf16_t* __restrict__ X, const bf16_t* __restrict__ W,
         const float* __restrict__ bias, float scale,
         bf16_t* __restrict__ Ybf, float* __restrict__ Yf) {
  int wave = threadIdx.x >> 5;
  int row0 = blockIdx.x * 128 + (wave & 3) * 32;
  int col0 = blockIdx.y * 64  + (wave >> 2) * 32;
  v8f acc[2][2] = {{{0.f}, {0.f}}, {{0.f}, {0.f}}};
  const bf16_t* Xa = X + (size_t)row0 * EMB;
  const bf16_t* Wb = W + (size_t)col0 * EMB;
#pragma unroll 4
  for (int k = 0; k < EMB; k += 32) {
    if (k + 128 < EMB) {
      __builtin_prefetch(Xa + k + 128, 0, 1);
      __builtin_prefetch(Wb + k + 128, 0, 1);
    }
    v16bf a0 = load_a_frag(Xa + k, EMB);
    v16bf a1 = load_a_frag(Xa + 16 * EMB + k, EMB);
    v16bf b0 = load_bT_frag(Wb + k, EMB);
    v16bf b1 = load_bT_frag(Wb + 16 * EMB + k, EMB);
    acc[0][0] = wmma_bf16(a0, b0, acc[0][0]);
    acc[0][1] = wmma_bf16(a0, b1, acc[0][1]);
    acc[1][0] = wmma_bf16(a1, b0, acc[1][0]);
    acc[1][1] = wmma_bf16(a1, b1, acc[1][1]);
  }
  int lane = threadIdx.x & 31, nn = lane & 15, half = lane >> 4;
#pragma unroll
  for (int mt = 0; mt < 2; ++mt)
#pragma unroll
    for (int nt = 0; nt < 2; ++nt)
#pragma unroll
      for (int r = 0; r < 8; ++r) {
        int row = row0 + mt * 16 + r + 8 * half;
        int col = col0 + nt * 16 + nn;
        float y = (acc[mt][nt][r] + bias[col]) * scale;
        if (F32OUT) Yf[(size_t)row * EMB + col] = y;
        else        Ybf[(size_t)row * EMB + col] = (bf16_t)y;
      }
}

// ---------------- pattn = softplus-like(kp_head @ pq_head^T) ---------------
// per (b,h): (2048 x 64) @ (64 x 32); f(x)=log2(1+2^x). Output bf16
// (NBH, 2048, 32). Block 256 (8 waves x 16 rows), grid (TGT/128, NBH).

__global__ void __launch_bounds__(256)
pattn_kernel(const bf16_t* __restrict__ kp, const bf16_t* __restrict__ pq,
             bf16_t* __restrict__ pattn) {
  int bh = blockIdx.y, b = bh >> 4, h = bh & 15;
  int wave = threadIdx.x >> 5;
  int row0 = blockIdx.x * 128 + wave * 16;
  const bf16_t* Ab = kp + ((size_t)row0 * BSZ + b) * EMB + h * HD;
  const bf16_t* Bb = pq + (size_t)b * EMB + h * HD;
  v8f acc[2] = {{0.f}, {0.f}};
#pragma unroll
  for (int k = 0; k < HD; k += 32) {
    v16bf a  = load_a_frag(Ab + k, BSZ * EMB);
    v16bf b0 = load_bT_frag(Bb + k, BSZ * EMB);
    v16bf b1 = load_bT_frag(Bb + (size_t)16 * BSZ * EMB + k, BSZ * EMB);
    acc[0] = wmma_bf16(a, b0, acc[0]);
    acc[1] = wmma_bf16(a, b1, acc[1]);
  }
  int lane = threadIdx.x & 31, nn = lane & 15, half = lane >> 4;
  bf16_t* outr = pattn + (size_t)bh * TGT * PLEN;
#pragma unroll
  for (int nt = 0; nt < 2; ++nt)
#pragma unroll
    for (int r = 0; r < 8; ++r) {
      int row = row0 + r + 8 * half;
      int p = nt * 16 + nn;
      float x = acc[nt][r];
      // softplus(beta*x)/beta with beta=ln2  ==  log2(1 + 2^x)
      // v_exp_f32 / v_log_f32 are base-2 on AMD — exact mapping.
      float f = (x > 20.f) ? x
                           : __builtin_amdgcn_logf(1.f + __builtin_amdgcn_exp2f(x));
      outr[(size_t)row * PLEN + p] = (bf16_t)f;
    }
}

// ---------------- fused chunked causal linear attention --------------------
// One block per (b,h). Chunk T=32. States S1 (64x32), S2 (32x64) kept f32 in
// LDS (precision of the long prefix sums lives here); intra-chunk dense GEMMs
// go through WMMA with LDS-staged transposes / re-fragmentation.

__global__ void __launch_bounds__(256)
causal_fused(const bf16_t* __restrict__ qg, const bf16_t* __restrict__ kvg,
             const bf16_t* __restrict__ pg, bf16_t* __restrict__ attn) {
  __shared__ __align__(64) float  S1[HD][PLEN];   // sum kv x pattn
  __shared__ __align__(64) float  S2[PLEN][HD];   // sum pattn x kv
  __shared__ __align__(64) float  Qs[32][HD];
  __shared__ __align__(64) float  Ks[32][HD];
  __shared__ __align__(64) float  Ps[32][PLEN];
  __shared__ __align__(64) bf16_t Zt[PLEN][32];   // pattn^T (chunk)
  __shared__ __align__(64) bf16_t Vt[HD][32];     // kv^T (chunk)
  __shared__ __align__(64) bf16_t QKm[32][32];    // masked Q K^T, bf16
  __shared__ __align__(64) float  AW[32][PLEN];   // attn weights (f32)
  __shared__ __align__(64) bf16_t AWbf[32][PLEN]; // softmaxed weights, bf16
  __shared__ __align__(64) bf16_t APm[32][32];    // masked A P^T, bf16
  __shared__ __align__(64) float  OUT[32][HD];

  int bh = blockIdx.x, b = bh >> 4, h = bh & 15;
  int t = threadIdx.x, wave = t >> 5, lane = t & 31, nn = lane & 15, half = lane >> 4;
  const int ldT = BSZ * EMB;  // time-row stride in token-major activations

  const bf16_t* qh  = qg  + (size_t)b * EMB + h * HD;
  const bf16_t* kvh = kvg + (size_t)b * EMB + h * HD;
  const bf16_t* ph  = pg  + (size_t)bh * TGT * PLEN;
  bf16_t* oh = attn + (size_t)b * EMB + h * HD;

  for (int i = t; i < HD * PLEN; i += 256) { (&S1[0][0])[i] = 0.f; (&S2[0][0])[i] = 0.f; }
  __syncthreads();

  for (int g = 0; g < TGT; g += 32) {
    // ---- P1: stage chunk operands ----
#pragma unroll
    for (int r = 0; r < 8; ++r) {
      int idx = t + r * 256, i = idx >> 6, d = idx & 63;
      bf16_t kv = kvh[(size_t)(g + i) * ldT + d];
      Qs[i][d] = (float)qh[(size_t)(g + i) * ldT + d];
      Ks[i][d] = (float)kv;
      Vt[d][i] = kv;
    }
#pragma unroll
    for (int r = 0; r < 4; ++r) {
      int idx = t + r * 256, i = idx >> 5, p = idx & 31;
      bf16_t pv = ph[(size_t)(g + i) * PLEN + p];
      Ps[i][p] = (float)pv;
      Zt[p][i] = pv;
    }
    __syncthreads();

    // ---- P2: AW = Q @ S1_prev (f32 VALU) ----
    {
      int i = t >> 3, pb = (t & 7) * 4;
      float a0 = 0, a1 = 0, a2 = 0, a3 = 0;
      for (int d = 0; d < HD; ++d) {
        float qv = Qs[i][d];
        a0 = fmaf(qv, S1[d][pb + 0], a0); a1 = fmaf(qv, S1[d][pb + 1], a1);
        a2 = fmaf(qv, S1[d][pb + 2], a2); a3 = fmaf(qv, S1[d][pb + 3], a3);
      }
      AW[i][pb] = a0; AW[i][pb + 1] = a1; AW[i][pb + 2] = a2; AW[i][pb + 3] = a3;
    }
    __syncthreads();

    // ---- P3: QK = Q K^T (WMMA), causal mask -> QKm ----
    if (wave < 4) {
      int mt = wave & 1, nt = wave >> 1;
      v8f c = {0.f};
#pragma unroll
      for (int k = 0; k < HD; k += 32) {
        v16bf a  = load_a_frag(qh + (size_t)(g + mt * 16) * ldT + k, ldT);
        v16bf bb = load_bT_frag(kvh + (size_t)(g + nt * 16) * ldT + k, ldT);
        c = wmma_bf16(a, bb, c);
      }
#pragma unroll
      for (int r = 0; r < 8; ++r) {
        int i = mt * 16 + r + 8 * half, j = nt * 16 + nn;
        QKm[i][j] = (bf16_t)((j <= i) ? c[r] : 0.f);
      }
    }
    __syncthreads();

    // ---- P4: AW += QKm @ Z (WMMA, B from staged Z^T) ----
    if (wave < 4) {
      int mt = wave & 1, nt = wave >> 1;
      v16bf a  = load_a_frag(&QKm[mt * 16][0], 32);
      v16bf bb = load_bT_frag(&Zt[nt * 16][0], 32);
      v8f c = {0.f};
      c = wmma_bf16(a, bb, c);
#pragma unroll
      for (int r = 0; r < 8; ++r) {
        int i = mt * 16 + r + 8 * half, p = nt * 16 + nn;
        AW[i][p] += c[r];
      }
    }
    __syncthreads();

    // ---- P5: length-divide + row softmax over 32 (wave 0, lane=row) ----
    if (t < 32) {
      int i = t;
      float inv = 1.f / (float)(g + i + 1);
      float v[PLEN], mx = -3.0e38f;
#pragma unroll
      for (int p = 0; p < PLEN; ++p) { v[p] = AW[i][p] * inv; mx = fmaxf(mx, v[p]); }
      float s = 0.f;
#pragma unroll
      for (int p = 0; p < PLEN; ++p) { v[p] = __expf(v[p] - mx); s += v[p]; }
      float rs = 1.f / s;
#pragma unroll
      for (int p = 0; p < PLEN; ++p) {
        float w = v[p] * rs;
        AW[i][p] = w;
        AWbf[i][p] = (bf16_t)w;
      }
    }
    __syncthreads();

    // ---- P6: OUT = AWn @ S2_prev (f32 VALU) ----
    {
      int i = t >> 3, db = (t & 7) * 8;
      float a[8] = {0, 0, 0, 0, 0, 0, 0, 0};
      for (int p = 0; p < PLEN; ++p) {
        float wv = AW[i][p];
#pragma unroll
        for (int x = 0; x < 8; ++x) a[x] = fmaf(wv, S2[p][db + x], a[x]);
      }
#pragma unroll
      for (int x = 0; x < 8; ++x) OUT[i][db + x] = a[x];
    }
    __syncthreads();

    // ---- P7: waves 0-3: AP = AWn P^T (WMMA) + mask; waves 4-7: S1 += K^T P
    if (wave < 4) {
      int mt = wave & 1, nt = wave >> 1;
      v16bf a  = load_a_frag(&AWbf[mt * 16][0], PLEN);
      v16bf bb = load_bT_frag(ph + (size_t)(g + nt * 16) * PLEN, PLEN);
      v8f c = {0.f};
      c = wmma_bf16(a, bb, c);
#pragma unroll
      for (int r = 0; r < 8; ++r) {
        int i = mt * 16 + r + 8 * half, j = nt * 16 + nn;
        APm[i][j] = (bf16_t)((j <= i) ? c[r] : 0.f);
      }
    } else {
      int tt = t - 128, dd = tt >> 1, pb = (tt & 1) * 16;
      for (int p = pb; p < pb + 16; ++p) {
        float s = 0.f;
        for (int i2 = 0; i2 < 32; ++i2) s = fmaf(Ks[i2][dd], Ps[i2][p], s);
        S1[dd][p] += s;
      }
    }
    __syncthreads();

    // ---- P8: OUT += APm @ V (WMMA, B from staged V^T), all 8 waves ----
    {
      int mt = wave & 1, nt = wave >> 1;  // nt 0..3
      v16bf a  = load_a_frag(&APm[mt * 16][0], 32);
      v16bf bb = load_bT_frag(&Vt[nt * 16][0], 32);
      v8f c = {0.f};
      c = wmma_bf16(a, bb, c);
#pragma unroll
      for (int r = 0; r < 8; ++r) {
        int i = mt * 16 + r + 8 * half, d = nt * 16 + nn;
        OUT[i][d] += c[r];
      }
    }
    __syncthreads();

    // ---- P9: finalize (length divide) -> global bf16; S2 += P^T K ----
#pragma unroll
    for (int r = 0; r < 8; ++r) {
      int idx = t + r * 256, i = idx >> 6, d = idx & 63;
      float inv = 1.f / (float)(g + i + 1);
      oh[(size_t)(g + i) * ldT + d] = (bf16_t)(OUT[i][d] * inv);
    }
#pragma unroll
    for (int r = 0; r < 8; ++r) {
      int idx = t + r * 256, p = idx >> 6, d = idx & 63;
      float s = 0.f;
      for (int i2 = 0; i2 < 32; ++i2) s = fmaf(Ps[i2][p], Ks[i2][d], s);
      S2[p][d] += s;
    }
    __syncthreads();
  }
}

// ---------------------------------------------------------------------------

extern "C" void kernel_launch(void* const* d_in, const int* in_sizes, int n_in,
                              void* d_out, int out_size, void* d_ws, size_t ws_size,
                              hipStream_t stream) {
  (void)in_sizes; (void)n_in; (void)out_size; (void)ws_size;
  const float* query  = (const float*)d_in[0];
  const float* pquery = (const float*)d_in[1];
  const float* Wpq    = (const float*)d_in[2];
  const float* bpq    = (const float*)d_in[3];
  const float* Wq     = (const float*)d_in[4];
  const float* bq     = (const float*)d_in[5];
  const float* Wpc    = (const float*)d_in[6];
  const float* bpc    = (const float*)d_in[7];
  const float* Wc     = (const float*)d_in[8];
  const float* bc     = (const float*)d_in[9];
  const float* Wo     = (const float*)d_in[10];
  const float* bo     = (const float*)d_in[11];
  float* out = (float*)d_out;

  char* ws = (char*)d_ws;
  size_t off = 0;
  auto alloc = [&](size_t bytes) -> void* {
    void* p = ws + off;
    off += (bytes + 255) & ~(size_t)255;
    return p;
  };
  bf16_t* Xbf      = (bf16_t*)alloc((size_t)NTOK * EMB * 2);
  bf16_t* Wpc_bf   = (bf16_t*)alloc((size_t)EMB * EMB * 2);
  bf16_t* Wq_bf    = (bf16_t*)alloc((size_t)EMB * EMB * 2);
  bf16_t* Wc_bf    = (bf16_t*)alloc((size_t)EMB * EMB * 2);
  bf16_t* Wo_bf    = (bf16_t*)alloc((size_t)EMB * EMB * 2);
  bf16_t* kp_bf    = (bf16_t*)alloc((size_t)NTOK * EMB * 2);
  bf16_t* q_bf     = (bf16_t*)alloc((size_t)NTOK * EMB * 2);
  bf16_t* kv_bf    = (bf16_t*)alloc((size_t)NTOK * EMB * 2);
  bf16_t* pq_bf    = (bf16_t*)alloc((size_t)PLEN * BSZ * EMB * 2);
  bf16_t* pattn_bf = (bf16_t*)alloc((size_t)NBH * TGT * PLEN * 2);
  bf16_t* attn_bf  = (bf16_t*)alloc((size_t)NTOK * EMB * 2);

  // 1. precision casts
  cvt_f32_bf16<<<(NTOK * EMB + 255) / 256, 256, 0, stream>>>(query, Xbf, NTOK * EMB);
  cvt_f32_bf16<<<(EMB * EMB + 255) / 256, 256, 0, stream>>>(Wpc, Wpc_bf, EMB * EMB);
  cvt_f32_bf16<<<(EMB * EMB + 255) / 256, 256, 0, stream>>>(Wq,  Wq_bf,  EMB * EMB);
  cvt_f32_bf16<<<(EMB * EMB + 255) / 256, 256, 0, stream>>>(Wc,  Wc_bf,  EMB * EMB);
  cvt_f32_bf16<<<(EMB * EMB + 255) / 256, 256, 0, stream>>>(Wo,  Wo_bf,  EMB * EMB);

  // 2. pq projection (tiny, f32)
  pq_proj<<<dim3(EMB / 256, PLEN * BSZ), 256, 0, stream>>>(pquery, Wpq, bpq, pq_bf);

  // 3. token projections (WMMA bf16 GEMMs)
  dim3 gg(NTOK / 128, EMB / 64);
  gemm_xwT<false><<<gg, 256, 0, stream>>>(Xbf, Wpc_bf, bpc, 1.0f,    kp_bf, nullptr);
  gemm_xwT<false><<<gg, 256, 0, stream>>>(Xbf, Wq_bf,  bq,  SCALING, q_bf,  nullptr);
  gemm_xwT<false><<<gg, 256, 0, stream>>>(Xbf, Wc_bf,  bc,  1.0f,    kv_bf, nullptr);

  // 4. pattn + softplus
  pattn_kernel<<<dim3(TGT / 128, NBH), 256, 0, stream>>>(kp_bf, pq_bf, pattn_bf);

  // 5. fused causal linear attention
  causal_fused<<<NBH, 256, 0, stream>>>(q_bf, kv_bf, pattn_bf, attn_bf);

  // 6. output projection -> f32 d_out
  gemm_xwT<true><<<gg, 256, 0, stream>>>(attn_bf, Wo_bf, bo, 1.0f, nullptr, out);
}
